// AttnBlock_36910948942194
// MI455X (gfx1250) — matrix operand order repaired and checked by reading, not compiled, not run on hardware
//
#include <hip/hip_runtime.h>
#include <stdint.h>

// ---------------------------------------------------------------- constants
#define T_SEQ   2048
#define DMODEL  1024
#define NHEADS  16
#define HDIM    64
#define BATCH   4
#define M_ROWS  (BATCH * T_SEQ)      // 8192

typedef __attribute__((ext_vector_type(16))) __bf16 bf16x16;
typedef __attribute__((ext_vector_type(8)))  float  floatx8;

union FragU { bf16x16 v; uint4 q[2]; };

// A-fragment (16x32 bf16): lane = row M, elements 0..7 -> K = kbase+h+0..7,
// elements 8..15 -> K = kbase+h+16..23, h = 8 for lanes 16..31.
__device__ __forceinline__ bf16x16 load_a_frag(const __bf16* row, int kbase, int lane) {
  const int h = (lane & 16) ? 8 : 0;
  FragU f;
  f.q[0] = *(const uint4*)(row + kbase + h);
  f.q[1] = *(const uint4*)(row + kbase + h + 16);
  return f.v;
}

// B-fragment (32x16 bf16): lane = col N, elements 0..7 -> K = kbase+h+0..7,
// elements 8..15 -> K = kbase+h+16..23, h = 8 for lanes 16..31.
__device__ __forceinline__ bf16x16 load_b_frag(const __bf16* row, int kbase, int lane) {
  const int h = (lane & 16) ? 8 : 0;
  const uint4* p = (const uint4*)(row + kbase + h);
  FragU f;
  f.q[0] = p[0];
  f.q[1] = p[2];
  return f.v;
}

__device__ __forceinline__ floatx8 wmma_bf16(bf16x16 a, bf16x16 b, floatx8 c) {
  return __builtin_amdgcn_wmma_f32_16x16x32_bf16(false, a, false, b, (short)0, c,
                                                 false, false);
}

// ---------------------------------------------------------------- fp32 -> bf16
__global__ void cvt_f32_to_bf16_k(const float* __restrict__ in,
                                  __bf16* __restrict__ out, int n) {
  int i = blockIdx.x * blockDim.x + threadIdx.x;
  if (i < n) out[i] = (__bf16)in[i];
}

// ---------------------------------------------------------------- generic GEMM
// C[m,n] = sum_k A[m,k] * W[n,k]   (A: MxK bf16 row-major, W: NxK bf16 row-major)
// mode 0: write bf16 at [b,h,t,64]        (Q/K projection)
// mode 1: write bf16 at [b,h,64,t]        (V projection, transposed)
// mode 2: write f32 row-major [M,N]       (final output projection)
// block = 128 threads (4 waves); each wave -> 16(M) x 64(N); block -> 64x64.
__global__ void gemm_bf16_wmma_k(const __bf16* __restrict__ A,
                                 const __bf16* __restrict__ W,
                                 void* __restrict__ out,
                                 int M, int N, int K, int mode) {
  const int lane = threadIdx.x & 31;
  const int wave = threadIdx.x >> 5;
  const int n0 = blockIdx.x * 64;
  const int m0 = blockIdx.y * 64 + wave * 16;
  const int mlane = lane & 15;

  const __bf16* arow = A + (size_t)(m0 + mlane) * K;
  const __bf16* brow0 = W + (size_t)(n0 + 0 * 16 + mlane) * K;
  const __bf16* brow1 = W + (size_t)(n0 + 1 * 16 + mlane) * K;
  const __bf16* brow2 = W + (size_t)(n0 + 2 * 16 + mlane) * K;
  const __bf16* brow3 = W + (size_t)(n0 + 3 * 16 + mlane) * K;

  floatx8 acc0 = {}, acc1 = {}, acc2 = {}, acc3 = {};
  for (int k = 0; k < K; k += 32) {
    __builtin_prefetch(arow + k + 128, 0, 1);
    bf16x16 a = load_a_frag(arow, k, lane);
    acc0 = wmma_bf16(a, load_b_frag(brow0, k, lane), acc0);
    acc1 = wmma_bf16(a, load_b_frag(brow1, k, lane), acc1);
    acc2 = wmma_bf16(a, load_b_frag(brow2, k, lane), acc2);
    acc3 = wmma_bf16(a, load_b_frag(brow3, k, lane), acc3);
  }

  floatx8 accs[4] = {acc0, acc1, acc2, acc3};
  const int mhi = (lane >> 4) << 3;   // +8 rows for upper lane half
#pragma unroll
  for (int s = 0; s < 4; ++s) {
#pragma unroll
    for (int r = 0; r < 8; ++r) {
      int m = m0 + r + mhi;
      int n = n0 + s * 16 + mlane;
      float v = accs[s][r];
      if (mode == 2) {
        ((float*)out)[(size_t)m * N + n] = v;
      } else {
        int b = m >> 11, t = m & (T_SEQ - 1);
        int h = n >> 6, dl = n & (HDIM - 1);
        size_t idx = (mode == 0)
            ? ((size_t)((b * NHEADS + h) * T_SEQ + t) * HDIM + dl)
            : ((size_t)((b * NHEADS + h) * HDIM + dl) * T_SEQ + t);
        ((__bf16*)out)[idx] = (__bf16)v;
      }
    }
  }
}

// ---------------------------------------------------------------- RoPE (Q and K)
// Layout [b,h,t,64]; pair (2i, 2i+1) rotated by angle t * 10000^(-2i/64).
__global__ void rope_bf16_k(__bf16* __restrict__ q, __bf16* __restrict__ k,
                            int npairs) {
  int i = blockIdx.x * blockDim.x + threadIdx.x;
  if (i >= npairs) return;
  int pair = i & 31;                 // 32 pairs per row of 64
  int t = (i >> 5) & (T_SEQ - 1);
  float ang = (float)t * __powf(10000.f, -(float)(2 * pair) * (1.0f / 64.0f));
  float s, c;
  __sincosf(ang, &s, &c);
  size_t base = (size_t)i * 2;
  float q0 = (float)q[base], q1 = (float)q[base + 1];
  q[base]     = (__bf16)(q0 * c - q1 * s);
  q[base + 1] = (__bf16)(q0 * s + q1 * c);
  float k0 = (float)k[base], k1 = (float)k[base + 1];
  k[base]     = (__bf16)(k0 * c - k1 * s);
  k[base + 1] = (__bf16)(k0 * s + k1 * c);
}

// ---------------------------------------------------------------- flash attention
// grid: (T/64, B*H); block = 128 (4 waves). Wave w owns query rows
// q0 = blockIdx.x*64 + w*16 .. +15 and streams causal 64-key tiles.
// Qb,Kb: [b,h,t,64] bf16; Vt: [b,h,64,t] bf16; Ctx out: [b,t,1024] bf16.
__global__ void attn_fa_wmma_k(const __bf16* __restrict__ Qb,
                               const __bf16* __restrict__ Kb,
                               const __bf16* __restrict__ Vt,
                               __bf16* __restrict__ Ctx) {
  __shared__ __bf16 smP[4][16 * 72];   // per-wave 16 x 64 P tile, stride 72
  const int lane = threadIdx.x & 31;
  const int wave = threadIdx.x >> 5;
  const int mlane = lane & 15;
  const int mhi = (lane >> 4) << 3;
  const int bh = blockIdx.y;
  const int q0 = blockIdx.x * 64 + wave * 16;

  const __bf16* qbase = Qb + (size_t)bh * T_SEQ * HDIM;
  const __bf16* kbase = Kb + (size_t)bh * T_SEQ * HDIM;
  const __bf16* vbase = Vt + (size_t)bh * HDIM * T_SEQ;

  const __bf16* qrow = qbase + (size_t)(q0 + mlane) * HDIM;
  bf16x16 aQ0 = load_a_frag(qrow, 0, lane);
  bf16x16 aQ1 = load_a_frag(qrow, 32, lane);

  floatx8 oAcc[4] = {};
  float rmax[8], rsum[8];
#pragma unroll
  for (int r = 0; r < 8; ++r) { rmax[r] = -3.0e38f; rsum[r] = 0.f; }

  __bf16* pbuf = &smP[wave][0];
  const int ktiles = blockIdx.x + 1;          // causal tile bound
  const float scale = 0.125f;                 // 1/sqrt(64)

  for (int kt = 0; kt < ktiles; ++kt) {
    const int k0 = kt * 64;

    // ---- S = Q * K^T  (16 x 64), 4 subtiles, 2 chained WMMAs each
    floatx8 sAcc[4];
#pragma unroll
    for (int s = 0; s < 4; ++s) {
      const __bf16* krow = kbase + (size_t)(k0 + s * 16 + mlane) * HDIM;
      floatx8 c = {};
      c = wmma_bf16(aQ0, load_b_frag(krow, 0, lane), c);
      c = wmma_bf16(aQ1, load_b_frag(krow, 32, lane), c);
      sAcc[s] = c;
    }

    // ---- scale + causal mask + online softmax
    float newmax[8];
#pragma unroll
    for (int r = 0; r < 8; ++r) {
      int m = q0 + r + mhi;
      float mv = rmax[r];
#pragma unroll
      for (int s = 0; s < 4; ++s) {
        int n = k0 + s * 16 + mlane;
        float sv = sAcc[s][r] * scale;
        sv = (n <= m) ? sv : -3.0e38f;
        sAcc[s][r] = sv;
        mv = fmaxf(mv, sv);
      }
      for (int off = 1; off < 16; off <<= 1)            // stay inside 16-lane half
        mv = fmaxf(mv, __shfl_xor(mv, off, 32));
      newmax[r] = mv;
    }

#pragma unroll
    for (int r = 0; r < 8; ++r) {
      float corr = __expf(rmax[r] - newmax[r]);
      rsum[r] *= corr;
#pragma unroll
      for (int s2 = 0; s2 < 4; ++s2) oAcc[s2][r] *= corr;
      float lsum = 0.f;
#pragma unroll
      for (int s = 0; s < 4; ++s) {
        float p = __expf(sAcc[s][r] - newmax[r]);
        lsum += p;
        pbuf[(r + mhi) * 72 + (s * 16 + mlane)] = (__bf16)p;  // C-layout -> LDS
      }
      for (int off = 1; off < 16; off <<= 1)
        lsum += __shfl_xor(lsum, off, 32);
      rsum[r] += lsum;
      rmax[r] = newmax[r];
    }

    // ---- read P back as A-fragments (per-wave LDS, same-wave DS ordering)
    const __bf16* prow = pbuf + mlane * 72;
    bf16x16 aP0 = load_a_frag(prow, 0, lane);
    bf16x16 aP1 = load_a_frag(prow, 32, lane);

    // ---- O += P * V   (V transposed: keys contiguous -> direct B-fragments)
#pragma unroll
    for (int s2 = 0; s2 < 4; ++s2) {
      const __bf16* vrow = vbase + (size_t)(s2 * 16 + mlane) * T_SEQ;
      oAcc[s2] = wmma_bf16(aP0, load_b_frag(vrow, k0, lane), oAcc[s2]);
      oAcc[s2] = wmma_bf16(aP1, load_b_frag(vrow, k0 + 32, lane), oAcc[s2]);
    }
  }

  // ---- normalize and write ctx as [b, t, h*64+d] bf16
  const int b = bh >> 4, h = bh & 15;
#pragma unroll
  for (int r = 0; r < 8; ++r) {
    float inv = 1.0f / rsum[r];
    int t = q0 + r + mhi;
#pragma unroll
    for (int s2 = 0; s2 < 4; ++s2) {
      int dd = s2 * 16 + mlane;
      Ctx[(size_t)(b * T_SEQ + t) * DMODEL + h * HDIM + dd] =
          (__bf16)(oAcc[s2][r] * inv);
    }
  }
}

// ---------------------------------------------------------------- launch
extern "C" void kernel_launch(void* const* d_in, const int* in_sizes, int n_in,
                              void* d_out, int out_size, void* d_ws, size_t ws_size,
                              hipStream_t stream) {
  (void)in_sizes; (void)n_in; (void)out_size; (void)ws_size;
  const float* x  = (const float*)d_in[0];
  const float* Wq = (const float*)d_in[1];
  const float* Wk = (const float*)d_in[2];
  const float* Wv = (const float*)d_in[3];
  const float* Wo = (const float*)d_in[4];

  const size_t XE = (size_t)M_ROWS * DMODEL;   // 8,388,608 elems
  const size_t WE = (size_t)DMODEL * DMODEL;   // 1,048,576 elems

  __bf16* xb  = (__bf16*)d_ws;
  __bf16* wqb = xb + XE;
  __bf16* wkb = wqb + WE;
  __bf16* wvb = wkb + WE;
  __bf16* wob = wvb + WE;
  __bf16* Qb  = wob + WE;   // [B,H,T,64]
  __bf16* Kb  = Qb + XE;    // [B,H,T,64]
  __bf16* Vt  = Kb + XE;    // [B,H,64,T]
  __bf16* Ctx = Vt + XE;    // [M, DMODEL]

  const int thr = 256;
  cvt_f32_to_bf16_k<<<dim3((unsigned)((XE + thr - 1) / thr)), thr, 0, stream>>>(x, xb, (int)XE);
  cvt_f32_to_bf16_k<<<dim3((unsigned)((WE + thr - 1) / thr)), thr, 0, stream>>>(Wq, wqb, (int)WE);
  cvt_f32_to_bf16_k<<<dim3((unsigned)((WE + thr - 1) / thr)), thr, 0, stream>>>(Wk, wkb, (int)WE);
  cvt_f32_to_bf16_k<<<dim3((unsigned)((WE + thr - 1) / thr)), thr, 0, stream>>>(Wv, wvb, (int)WE);
  cvt_f32_to_bf16_k<<<dim3((unsigned)((WE + thr - 1) / thr)), thr, 0, stream>>>(Wo, wob, (int)WE);

  dim3 ggrid(DMODEL / 64, M_ROWS / 64);        // (16, 128)
  gemm_bf16_wmma_k<<<ggrid, 128, 0, stream>>>(xb, wqb, Qb, M_ROWS, DMODEL, DMODEL, 0);
  gemm_bf16_wmma_k<<<ggrid, 128, 0, stream>>>(xb, wkb, Kb, M_ROWS, DMODEL, DMODEL, 0);
  gemm_bf16_wmma_k<<<ggrid, 128, 0, stream>>>(xb, wvb, Vt, M_ROWS, DMODEL, DMODEL, 1);

  const int npairs = BATCH * NHEADS * T_SEQ * (HDIM / 2);   // 4,194,304
  rope_bf16_k<<<dim3((npairs + thr - 1) / thr), thr, 0, stream>>>(Qb, Kb, npairs);

  attn_fa_wmma_k<<<dim3(T_SEQ / 64, BATCH * NHEADS), 128, 0, stream>>>(Qb, Kb, Vt, Ctx);

  gemm_bf16_wmma_k<<<ggrid, 128, 0, stream>>>(Ctx, wob, d_out, M_ROWS, DMODEL, DMODEL, 2);
}
